// TaylorShiftAttention_14001593385544
// MI455X (gfx1250) — matrix operand order repaired and checked by reading, not compiled, not run on hardware
//
#include <hip/hip_runtime.h>
#include <hip/hip_bf16.h>
#include <stdint.h>

typedef __attribute__((ext_vector_type(16))) _Float16 v16h;
typedef __attribute__((ext_vector_type(8)))  _Float16 v8h;
typedef __attribute__((ext_vector_type(4)))  _Float16 v4h;
typedef __attribute__((ext_vector_type(8)))  float    v8f;

#define HD     8
#define NTOK   4096
#define CDIM   384
#define DHEAD  48
#define DPAD   64
#define QKV3   1152
#define NKV    49              // 48 kvmod slices + 1 plain kv slice
#define SLICE  (DPAD * DPAD)   // 4096 halfs per kvmod slice
#define VTILE  2048            // halfs per fragment-major v1 tile (32 n x 64 f)

#define SCALE      2.6321480259049848f   // 48^0.25
#define SB2        6.9282032302755088f   // scale^2 * B_COEF
#define SC4        48.0f                 // scale^4 * C_COEF
#define A_COEF     0.5f
#define CINV_SQRT  0.0510310363079829f   // 384^-0.5
#define INV_N      (1.0f/4096.0f)
#define EPS_N      1e-12f

// ---------- fragment-major swizzle helpers (CDNA5 ISA 7.12.2 layouts) ------
// A 16x32 (MxK): lane = m + 16*((k>>3)&1); aligned 8-groups along k contiguous.
__device__ __forceinline__ int a_swz(int m, int k) {
  return (m + 16 * ((k >> 3) & 1)) * 16 + ((k & 16) ? 8 : 0) + (k & 7);
}
// B 32x16 (KxN): lane = n + 16*((k>>3)&1); aligned 8-groups along k contiguous.
__device__ __forceinline__ int b_swz(int k, int n) {
  return ((n & 15) + 16 * ((k >> 3) & 1)) * 16 + ((k & 16) ? 8 : 0) + (k & 7);
}
// contiguous 32B fragment read (2x ds_load_b128)
__device__ __forceinline__ v16h frag_ld(const _Float16* tile, int lane) {
  return *(const v16h*)(tile + lane * 16);
}
// scalar (row-major) A fragment load, for tiles that must stay row-major
__device__ __forceinline__ v16h frag_a_rowmajor(const _Float16* As, int lda, int lane) {
  const int m = lane & 15;
  const int kb = (lane < 16) ? 0 : 8;
  v16h a;
#pragma unroll
  for (int p = 0; p < 8; ++p) {
    const int k = (p < 4) ? (kb + 2 * p) : (16 + kb + 2 * (p - 4));
    a[2 * p]     = As[m * lda + k];
    a[2 * p + 1] = As[m * lda + k + 1];
  }
  return a;
}

// ---------- async global->LDS copies (ASYNCcnt-tracked) --------------------
__device__ __forceinline__ void async_b128(const _Float16* gsrc, unsigned lds_byte) {
  asm volatile("global_load_async_to_lds_b128 %0, %1, off"
               :: "v"(lds_byte), "v"(gsrc) : "memory");
}
// 4096 halfs (8 KB): 4 instructions/thread
__device__ __forceinline__ void async_slice8k(const _Float16* gsrc,
                                              _Float16* ldst, int tid) {
  const unsigned lbase = (unsigned)(unsigned long long)ldst;
#pragma unroll
  for (int j = 0; j < 4; ++j) {
    const int t = tid + j * 128;
    async_b128(gsrc + t * 8, lbase + t * 16);
  }
}
// 2048 halfs (4 KB): 2 instructions/thread
__device__ __forceinline__ void async_tile4k(const _Float16* gsrc,
                                             _Float16* ldst, int tid) {
  const unsigned lbase = (unsigned)(unsigned long long)ldst;
#pragma unroll
  for (int j = 0; j < 2; ++j) {
    const int t = tid + j * 128;
    async_b128(gsrc + t * 8, lbase + t * 16);
  }
}

// ---------------- Kernels 1/6: Out[M,N] = A[M,K] * Bw[N,K]^T + bias --------

template <int N, int K>
__global__ __launch_bounds__(128)
void wmma_gemm_xt(const float* __restrict__ A, const float* __restrict__ Bw,
                  const float* __restrict__ bias, float* __restrict__ Out) {
  __shared__ __align__(32) _Float16 As[512];        // 16x32 tile, frag-major
  __shared__ __align__(32) _Float16 Bs[4 * 512];    // four 32x16 tiles, frag-major
  const int tid  = threadIdx.x;
  const int lane = tid & 31;
  const int w    = tid >> 5;
  const int m0   = blockIdx.y * 16;
  const int n0   = blockIdx.x * 64;
  v8f acc = {};
  for (int k0 = 0; k0 < K; k0 += 32) {
    {                                                // A tile: 1 float4 / thread
      const int m = tid >> 3, k4 = (tid & 7) * 4;
      const float4 f = *(const float4*)(A + (size_t)(m0 + m) * K + k0 + k4);
      const v4h hv = {(_Float16)f.x, (_Float16)f.y, (_Float16)f.z, (_Float16)f.w};
      *(v4h*)(As + a_swz(m, k4)) = hv;
    }
#pragma unroll
    for (int g = tid; g < 512; g += 128) {           // B tiles: 4 float4 / thread
      const int nl = g >> 3, k4 = (g & 7) * 4;
      const float4 f = *(const float4*)(Bw + (size_t)(n0 + nl) * K + k0 + k4);
      const v4h hv = {(_Float16)f.x, (_Float16)f.y, (_Float16)f.z, (_Float16)f.w};
      *(v4h*)(Bs + (nl >> 4) * 512 + b_swz(k4, nl & 15)) = hv;
    }
    __syncthreads();
    const v16h a = frag_ld(As, lane);
    const v16h b = frag_ld(Bs + w * 512, lane);
    acc = __builtin_amdgcn_wmma_f32_16x16x32_f16(false, a, false, b,
                                                 (short)0, acc, false, false);
    __syncthreads();
  }
  const int ncol = n0 + 16 * w + (lane & 15);
  const float bv = bias[ncol];
#pragma unroll
  for (int v = 0; v < 8; ++v) {
    const int m = v + ((lane < 16) ? 0 : 8);
    Out[(size_t)(m0 + m) * N + ncol] = acc[v] + bv;
  }
}

// ---------------- Kernel 2: split qkv, normalize, pack f16 -----------------
// KhT row 63 is set to 1.0 (the "ones row" used by the plain-kv slice).
// V1 is written fragment-major per 32-token tile for linear async staging.

__global__ __launch_bounds__(256)
void norm_pack_kernel(const float* __restrict__ qkv, const float* __restrict__ temp,
                      _Float16* __restrict__ Qh, _Float16* __restrict__ KhT,
                      _Float16* __restrict__ V1s) {
  const int idx = blockIdx.x * 256 + threadIdx.x;   // h*4096 + n
  const int h = idx >> 12;
  const int n = idx & 4095;
  const float* qr = qkv + (size_t)n * QKV3 + h * DHEAD;
  const float* kr = qr + CDIM;
  const float* vr = qr + 2 * CDIM;
  float qn2 = 0.f, kn2 = 0.f;
#pragma unroll 4
  for (int dd = 0; dd < DHEAD; ++dd) {
    const float qv = qr[dd] * CINV_SQRT;  qn2 += qv * qv;
    const float kv = kr[dd];              kn2 += kv * kv;
  }
  const float qs = SCALE * temp[h] / fmaxf(sqrtf(qn2), EPS_N);
  const float ks = SCALE / fmaxf(sqrtf(kn2), EPS_N);
  const size_t rowq = ((size_t)h * NTOK + n) * DPAD;
  const size_t vtile = ((size_t)h * 128 + (n >> 5)) * VTILE;
  const int nn = n & 31;
#pragma unroll 4
  for (int dd = 0; dd < DPAD; ++dd) {
    const float qo = (dd < DHEAD) ? qr[dd] * CINV_SQRT * qs : 0.f;
    const float ko = (dd < DHEAD) ? kr[dd] * ks : ((dd == DPAD - 1) ? 1.f : 0.f);
    const float vo = (dd == 0) ? INV_N : ((dd <= DHEAD) ? vr[dd - 1] * INV_N : 0.f);
    Qh[rowq + dd] = (_Float16)qo;
    KhT[((size_t)h * DPAD + dd) * NTOK + n] = (_Float16)ko;
    V1s[vtile + (dd >> 4) * 512 + b_swz(nn, dd & 15)] = (_Float16)vo;
  }
}

// ---------------- Kernel 3: v1 column sums (C-term) ------------------------

__global__ __launch_bounds__(64)
void v1sum_kernel(const _Float16* __restrict__ V1s, float* __restrict__ v1sum) {
  const int h = blockIdx.x, f = threadIdx.x;
  float s = 0.f;
  for (int n = 0; n < NTOK; ++n)
    s += (float)V1s[((size_t)h * 128 + (n >> 5)) * VTILE +
                    (f >> 4) * 512 + b_swz(n & 31, f & 15)];
  v1sum[h * DPAD + f] = s;
}

// ------- Kernel 4: kvmod[i] = (K ⊙ k_i)^T @ V1 ; i=48 -> plain kv ----------
// B (v1) tiles arrive via async double buffer; A tiles staged with packed
// f16 multiplies; output stored B-fragment-major for kernel 5.

__global__ __launch_bounds__(128)
void kvmod_kernel(const _Float16* __restrict__ KhT, const _Float16* __restrict__ V1s,
                  _Float16* __restrict__ kvmodh) {
  __shared__ __align__(32) _Float16 Asc[4 * 512];     // 64x32 scaled k^T, frag-major
  __shared__ __align__(32) _Float16 Vs[2][VTILE];     // double-buffered v1 tiles
  const int i = blockIdx.x;           // 0..48
  const int h = blockIdx.y;
  const int tid = threadIdx.x, lane = tid & 31, w = tid >> 5;   // wave -> j-tile
  const int i_src = (i < DHEAD) ? i : (DPAD - 1);     // ones row for plain kv
  const _Float16* krow_s = KhT + ((size_t)h * DPAD + i_src) * NTOK;
  const _Float16* vbase  = V1s + (size_t)h * 128 * VTILE;
  v8f acc[4] = {{}, {}, {}, {}};

  async_tile4k(vbase, Vs[0], tid);                    // prime
  for (int c = 0; c < 128; ++c) {
    const int cur = c & 1;
    if (c + 1 < 128) {
      async_tile4k(vbase + (size_t)(c + 1) * VTILE, Vs[1 - cur], tid);
      asm volatile("s_wait_asynccnt 0x2" ::: "memory");   // tile c landed
    } else {
      asm volatile("s_wait_asynccnt 0x0" ::: "memory");
    }
    const int n0 = c * 32;
#pragma unroll
    for (int g = tid; g < 256; g += 128) {            // A tile: 8-half groups
      const int j = g >> 2, nng = (g & 3) * 8;
      const v8h kv8 = *(const v8h*)(KhT + ((size_t)h * DPAD + j) * NTOK + n0 + nng);
      const v8h si8 = *(const v8h*)(krow_s + n0 + nng);
      *(v8h*)(Asc + (j >> 4) * 512 +
              ((j & 15) + 16 * ((nng >> 3) & 1)) * 16 + ((nng & 16) ? 8 : 0)) =
          kv8 * si8;
    }
    __syncthreads();
    const v16h a = frag_ld(Asc + w * 512, lane);
#pragma unroll
    for (int ft = 0; ft < 4; ++ft) {
      const v16h b = frag_ld(Vs[cur] + ft * 512, lane);
      acc[ft] = __builtin_amdgcn_wmma_f32_16x16x32_f16(false, a, false, b,
                                                       (short)0, acc[ft], false, false);
    }
    __syncthreads();
  }
  const size_t base = ((size_t)h * NKV + i) * SLICE;
#pragma unroll
  for (int ft = 0; ft < 4; ++ft) {
#pragma unroll
    for (int v = 0; v < 8; ++v) {
      const int j = 16 * w + v + ((lane < 16) ? 0 : 8);   // K index of B
      const int f = 16 * ft + (lane & 15);                // N index of B
      kvmodh[base + ((j >> 5) * 4 + ft) * 512 + b_swz(j & 31, f & 15)] =
          (_Float16)acc[ft][v];
    }
  }
}

// ---------------- Kernel 5: y accumulation (async-pipelined), divide -------

__global__ __launch_bounds__(128)
void y_kernel(const _Float16* __restrict__ Qh, const _Float16* __restrict__ kvmodh,
              const float* __restrict__ v1sum, float* __restrict__ attn) {
  __shared__ __align__(32) _Float16 Qs[16 * 64];       // row-major (scalar reuse)
  __shared__ __align__(32) _Float16 Bsc[2][SLICE];     // double-buffered slices
  __shared__ float ynorm[16];
  const int n0 = blockIdx.x * 16;
  const int h  = blockIdx.y;
  const int tid = threadIdx.x, lane = tid & 31, w = tid >> 5;   // wave -> f-tile
  const _Float16* slice0 = kvmodh + (size_t)h * NKV * SLICE;
  const _Float16* qsrc   = Qh + ((size_t)h * NTOK + n0) * DPAD; // 1024 contiguous

  async_b128(qsrc + tid * 8, (unsigned)(unsigned long long)Qs + tid * 16);
  async_slice8k(slice0, Bsc[0], tid);
  asm volatile("s_wait_asynccnt 0x4" ::: "memory");    // Qs landed (in-order)
  __syncthreads();
  const v16h a0 = frag_a_rowmajor(Qs, 64, lane);       // d 0..31 (once per block)
  const v16h a1 = frag_a_rowmajor(Qs + 32, 64, lane);  // d 32..63

  v8f acc = {};
  for (int i = 0; i < NKV; ++i) {
    const int cur = i & 1;
    if (i + 1 < NKV) {
      async_slice8k(slice0 + (size_t)(i + 1) * SLICE, Bsc[1 - cur], tid);
      asm volatile("s_wait_asynccnt 0x4" ::: "memory");  // slice i landed
    } else {
      asm volatile("s_wait_asynccnt 0x0" ::: "memory");
    }
    __syncthreads();
    const _Float16* B = Bsc[cur];
    const v16h b0 = frag_ld(B + w * 512, lane);          // K chunk 0
    const v16h b1 = frag_ld(B + (4 + w) * 512, lane);    // K chunk 1
    v8f tmp = {};
    tmp = __builtin_amdgcn_wmma_f32_16x16x32_f16(false, a0, false, b0,
                                                 (short)0, tmp, false, false);
    tmp = __builtin_amdgcn_wmma_f32_16x16x32_f16(false, a1, false, b1,
                                                 (short)0, tmp, false, false);
#pragma unroll
    for (int v = 0; v < 8; ++v) {
      const int m = v + ((lane < 16) ? 0 : 8);
      const float rs = (i < DHEAD) ? (A_COEF * (float)Qs[m * 64 + i]) : SB2;
      acc[v] += rs * tmp[v];
    }
    __syncthreads();
  }
  const int f = 16 * w + (lane & 15);
  const float cterm = SC4 * v1sum[h * DPAD + f];
#pragma unroll
  for (int v = 0; v < 8; ++v) acc[v] += cterm;
  if (w == 0 && (lane & 15) == 0) {
#pragma unroll
    for (int v = 0; v < 8; ++v) ynorm[v + ((lane < 16) ? 0 : 8)] = acc[v];
  }
  __syncthreads();
  if (f >= 1 && f <= DHEAD) {
#pragma unroll
    for (int v = 0; v < 8; ++v) {
      const int m = v + ((lane < 16) ? 0 : 8);
      attn[(size_t)(n0 + m) * CDIM + h * DHEAD + (f - 1)] = acc[v] / ynorm[m];
    }
  }
}

// ---------------- host launcher --------------------------------------------

extern "C" void kernel_launch(void* const* d_in, const int* in_sizes, int n_in,
                              void* d_out, int out_size, void* d_ws, size_t ws_size,
                              hipStream_t stream) {
  const float* x      = (const float*)d_in[0];
  const float* qkv_w  = (const float*)d_in[1];
  const float* qkv_b  = (const float*)d_in[2];
  const float* proj_w = (const float*)d_in[3];
  const float* proj_b = (const float*)d_in[4];
  const float* temp   = (const float*)d_in[5];

  char* p = (char*)d_ws;
  float*    qkv_raw = (float*)p;      p += (size_t)NTOK * QKV3 * 4;            // 18.9 MB
  _Float16* Qh      = (_Float16*)p;   p += (size_t)HD * NTOK * DPAD * 2;       // 4.2 MB
  _Float16* KhT     = (_Float16*)p;   p += (size_t)HD * DPAD * NTOK * 2;       // 4.2 MB
  _Float16* V1s     = (_Float16*)p;   p += (size_t)HD * NTOK * DPAD * 2;       // 4.2 MB
  _Float16* kvmodh  = (_Float16*)p;   p += (size_t)HD * NKV * SLICE * 2;       // 3.2 MB
  float*    v1sum   = (float*)p;      p += (size_t)HD * DPAD * 4;
  float*    attn    = (float*)p;      // 6.3 MB

  wmma_gemm_xt<QKV3, CDIM><<<dim3(QKV3 / 64, NTOK / 16), 128, 0, stream>>>(
      x, qkv_w, qkv_b, qkv_raw);
  norm_pack_kernel<<<dim3((HD * NTOK) / 256), 256, 0, stream>>>(
      qkv_raw, temp, Qh, KhT, V1s);
  v1sum_kernel<<<dim3(HD), 64, 0, stream>>>(V1s, v1sum);
  kvmod_kernel<<<dim3(NKV, HD), 128, 0, stream>>>(KhT, V1s, kvmodh);
  y_kernel<<<dim3(NTOK / 16, HD), 128, 0, stream>>>(Qh, kvmodh, v1sum, attn);
  wmma_gemm_xt<CDIM, CDIM><<<dim3(CDIM / 64, NTOK / 16), 128, 0, stream>>>(
      attn, proj_w, proj_b, (float*)d_out);
}
